// GNN_24799141167466
// MI455X (gfx1250) — matrix operand order repaired and checked by reading, not compiled, not run on hardware
//
#include <hip/hip_runtime.h>
#include <hip/hip_bf16.h>
#include <stdint.h>

// ---------------------------------------------------------------------------
// 2-layer GCN for MI455X (gfx1250, wave32).
// GEMMs run on v_wmma_f32_16x16x32_bf16; edge aggregation uses
// global_atomic_add_f32; weights staged transposed in LDS as bf16.
// ---------------------------------------------------------------------------

typedef __attribute__((ext_vector_type(16))) __bf16 v16bf;
typedef __attribute__((ext_vector_type(8)))  float  v8f;

#define D_IN      128
#define D_HID     128
#define N_CLASSES 32

__device__ __forceinline__ uint32_t hash32(uint32_t x) {
    x ^= x >> 17; x *= 0xed5ad4bbu;
    x ^= x >> 11; x *= 0xac4c1b51u;
    x ^= x >> 15; x *= 0x31848babu;
    x ^= x >> 14;
    return x;
}

// deg[i] = 1.0f (self-loop pre-counted)
__global__ void k_init_deg(float* __restrict__ deg, int n) {
    int i = blockIdx.x * blockDim.x + threadIdx.x;
    if (i < n) deg[i] = 1.0f;
}

// deg[dst[e]] += 1
__global__ void k_deg_accum(const int* __restrict__ dst, float* __restrict__ deg, int e) {
    int i = blockIdx.x * blockDim.x + threadIdx.x;
    if (i < e) atomicAdd(&deg[dst[i]], 1.0f);
}

// deg -> rsqrt(deg) in place
__global__ void k_dinv(float* __restrict__ deg, int n) {
    int i = blockIdx.x * blockDim.x + threadIdx.x;
    if (i < n) deg[i] = rsqrtf(deg[i]);
}

__global__ void k_zero(float* __restrict__ p, int n) {
    int i = blockIdx.x * blockDim.x + threadIdx.x;
    if (i < n) p[i] = 0.0f;
}

// ---------------------------------------------------------------------------
// GEMM: Out[nrows x NOUT] = A[nrows x 128] @ W[128 x NOUT], bf16 WMMA.
// Block = 256 threads = 8 waves; one wave owns a 16-row strip across all
// NOUT/16 column tiles. K = 128 -> 4 chunks of 32. W is staged transposed
// ([n][k]) in LDS as bf16 so B-fragment pairs (k, k+1) are one ds_load_b32.
// ---------------------------------------------------------------------------
template <int NOUT>
__launch_bounds__(256)
__global__ void k_gemm_wmma(const float* __restrict__ A, const float* __restrict__ W,
                            float* __restrict__ Out, int nrows) {
    __shared__ __bf16 sWt[NOUT * 128];           // [n][k], 32 KB (128) / 8 KB (32)

    const int tid = threadIdx.x;
    for (int idx = tid; idx < NOUT * 128; idx += 256) {   // coalesced read of W
        int k = idx / NOUT, n = idx % NOUT;
        sWt[n * 128 + k] = (__bf16)W[idx];
    }
    __syncthreads();

    const int wave = tid >> 5;
    const int lane = tid & 31;
    const int half = lane >> 4;      // 0: lanes 0-15, 1: lanes 16-31
    const int l16  = lane & 15;
    const int mtile = blockIdx.x * 8 + wave;     // wave-uniform
    if (mtile * 16 >= nrows) return;             // uniform branch: EXEC stays full
    const int m0 = mtile * 16;

    const float* arow = A + (size_t)(m0 + l16) * 128;   // A row for this lane

    constexpr int NT = NOUT / 16;
    v8f acc[NT];
    v8f vz = {};
#pragma unroll
    for (int nt = 0; nt < NT; ++nt) acc[nt] = vz;

#pragma unroll
    for (int kk = 0; kk < 4; ++kk) {             // K chunks of 32
        // A fragment (16x32 bf16): VGPR v<4 -> K = 8*half + 2v(+1);
        //                          VGPR v>=4 -> +16 block.
        v16bf a;
#pragma unroll
        for (int v = 0; v < 8; ++v) {
            int kb = kk * 32 + ((v < 4) ? 0 : 16) + half * 8 + (v & 3) * 2;
            float2 t = *(const float2*)(arow + kb);
            a[2 * v]     = (__bf16)t.x;
            a[2 * v + 1] = (__bf16)t.y;
        }
#pragma unroll
        for (int nt = 0; nt < NT; ++nt) {
            // B fragment (32x16 bf16): lane = col, VGPR v<4 -> K = 8*half + 2v(+1); v>=4 -> +16 block
            v16bf b;
            int n = nt * 16 + l16;
#pragma unroll
            for (int v = 0; v < 8; ++v) {
                int k = kk * 32 + ((v < 4) ? 0 : 16) + half * 8 + (v & 3) * 2;
                b[2 * v]     = sWt[n * 128 + k];
                b[2 * v + 1] = sWt[n * 128 + k + 1];
            }
            acc[nt] = __builtin_amdgcn_wmma_f32_16x16x32_bf16(
                false, a, false, b, (short)0, acc[nt], false, false);
        }
    }

    // D (16x16 f32): VGPR v holds row v + 8*half, col = lane&15
#pragma unroll
    for (int nt = 0; nt < NT; ++nt) {
#pragma unroll
        for (int v = 0; v < 8; ++v) {
            int row = m0 + v + 8 * half;
            Out[(size_t)row * NOUT + nt * 16 + l16] = acc[nt][v];
        }
    }
}

// ---------------------------------------------------------------------------
// Edge aggregation: out[dst] += h[src] * dinv[src]*dinv[dst]; virtual edges
// [E, E+N) are the self-loops. 256/F edges per block, F feature lanes each.
// ---------------------------------------------------------------------------
template <int F>
__launch_bounds__(256)
__global__ void k_aggregate(const float* __restrict__ h, const float* __restrict__ dinv,
                            const int* __restrict__ src, const int* __restrict__ dst,
                            float* __restrict__ out, int nedges, int nnodes) {
    constexpr int GPB = 256 / F;
    int g = blockIdx.x * GPB + (int)(threadIdx.x / F);
    int f = threadIdx.x & (F - 1);
    if (g >= nedges + nnodes) return;
    int s, d; float nrm;
    if (g < nedges) {
        s = src[g]; d = dst[g];
        nrm = dinv[s] * dinv[d];
    } else {
        s = d = g - nedges;
        float di = dinv[s];
        nrm = di * di;
    }
    atomicAdd(&out[(size_t)d * F + f], h[(size_t)s * F + f] * nrm);
}

// h = dropout(relu(h + b1)), p=0.5, deterministic hash mask, scale 2x
__global__ void k_bias_relu_dropout(float* __restrict__ h, const float* __restrict__ b,
                                    int total) {
    int i = blockIdx.x * blockDim.x + threadIdx.x;
    if (i >= total) return;
    float v = fmaxf(h[i] + b[i & (D_HID - 1)], 0.0f);
    uint32_t r = hash32((uint32_t)i ^ 0x9e3779b9u);
    h[i] = (r & 1u) ? (v * 2.0f) : 0.0f;
}

// log_softmax over 32 classes: one wave32 per node, shuffle reductions
__launch_bounds__(256)
__global__ void k_logsoftmax(float* __restrict__ out, const float* __restrict__ b2,
                             int nnodes) {
    int wave = threadIdx.x >> 5;
    int lane = threadIdx.x & 31;
    int node = blockIdx.x * 8 + wave;
    if (node >= nnodes) return;                  // wave-uniform
    float v = out[(size_t)node * N_CLASSES + lane] + b2[lane];
    float m = v;
#pragma unroll
    for (int off = 16; off >= 1; off >>= 1) m = fmaxf(m, __shfl_xor(m, off, 32));
    float s = __expf(v - m);
#pragma unroll
    for (int off = 16; off >= 1; off >>= 1) s += __shfl_xor(s, off, 32);
    out[(size_t)node * N_CLASSES + lane] = v - m - __logf(s);
}

// ---------------------------------------------------------------------------
extern "C" void kernel_launch(void* const* d_in, const int* in_sizes, int n_in,
                              void* d_out, int out_size, void* d_ws, size_t ws_size,
                              hipStream_t stream) {
    const float* x  = (const float*)d_in[0];
    const float* W1 = (const float*)d_in[1];
    const float* b1 = (const float*)d_in[2];
    const float* W2 = (const float*)d_in[3];
    const float* b2 = (const float*)d_in[4];
    const int*   ei = (const int*)d_in[5];

    const int N = in_sizes[0] / D_IN;     // 50000
    const int E = in_sizes[5] / 2;        // 800000
    const int* src = ei;
    const int* dst = ei + E;

    // workspace layout (floats): dinv[N] | h1[N*128] | agg1[N*128] | h2[N*32]
    float* dinv = (float*)d_ws;
    float* h1   = dinv + N;
    float* agg1 = h1 + (size_t)N * D_HID;
    float* h2   = agg1 + (size_t)N * D_HID;
    float* outp = (float*)d_out;

    const int T = 256;
    const int totE = E + N;               // real edges + self-loops

    // --- normalization ---
    k_init_deg <<<(N + T - 1) / T, T, 0, stream>>>(dinv, N);
    k_deg_accum<<<(E + T - 1) / T, T, 0, stream>>>(dst, dinv, E);
    k_dinv     <<<(N + T - 1) / T, T, 0, stream>>>(dinv, N);

    // --- layer 1: h1 = x @ W1 (WMMA), aggregate, bias+relu+dropout ---
    k_gemm_wmma<D_HID><<<(N / 16 + 7) / 8, T, 0, stream>>>(x, W1, h1, N);
    k_zero<<<((N * D_HID) + T - 1) / T, T, 0, stream>>>(agg1, N * D_HID);
    k_aggregate<D_HID><<<(totE + 1) / 2, T, 0, stream>>>(h1, dinv, src, dst, agg1, E, N);
    k_bias_relu_dropout<<<((N * D_HID) + T - 1) / T, T, 0, stream>>>(agg1, b1, N * D_HID);

    // --- layer 2: h2 = agg1 @ W2 (WMMA), aggregate into d_out, log_softmax ---
    k_gemm_wmma<N_CLASSES><<<(N / 16 + 7) / 8, T, 0, stream>>>(agg1, W2, h2, N);
    k_zero<<<((N * N_CLASSES) + T - 1) / T, T, 0, stream>>>(outp, N * N_CLASSES);
    k_aggregate<N_CLASSES><<<(totE + 7) / 8, T, 0, stream>>>(h2, dinv, src, dst, outp, E, N);
    k_logsoftmax<<<(N + 7) / 8, T, 0, stream>>>(outp, b2, N);
}
